// DeepARNetwork_81724637708851
// MI455X (gfx1250) — compile-verified, matrix operand order checked by reading, not run on hardware
//
#include <hip/hip_runtime.h>
#include <hip/hip_bf16.h>
#include <math.h>

// Problem constants (from setup_inputs)
#define B_    4096
#define C_    256
#define H_    24
#define T_    280   // C_ + H_
#define F_    5
#define E_    10
#define HID_  64
#define G4_   256   // 4*HID
#define NLAG  5

typedef __attribute__((ext_vector_type(16))) _Float16 v16h;
typedef __attribute__((ext_vector_type(8)))  float    v8f;

__device__ __constant__ int c_lags[NLAG] = {1, 2, 3, 7, 14};

// ---------------------------------------------------------------------------
// helpers
// ---------------------------------------------------------------------------
__device__ __forceinline__ float fast_sigmoid(float x) {
  return __builtin_amdgcn_rcpf(1.f + __expf(-x));
}
__device__ __forceinline__ float fast_tanh(float x) {
#if __has_builtin(__builtin_amdgcn_tanhf)
  return __builtin_amdgcn_tanhf(x);
#else
  return tanhf(x);
#endif
}
__device__ __forceinline__ float softplusf(float x) {
  return (x > 20.f) ? x : log1pf(__expf(x));
}

// Load a 16x32 f16 A-fragment (or equivalently a B-fragment of W^T given
// row-major W (NxK)) from LDS. Layout per CDNA5 ISA 7.12.2:
//   lane<16 : row l, K = {0..7, 16..23};  lane>=16 : row l, K = {8..15, 24..31}
__device__ __forceinline__ v16h ld_frag16(const _Float16* p, int stride) {
  const int lane = threadIdx.x & 31;
  const int l = lane & 15, hb = lane >> 4;
  const _Float16* r0 = p + l * stride + hb * 8;
  union { v16h v; struct { float4 a, b; } q; } u;
  u.q.a = *reinterpret_cast<const float4*>(r0);
  u.q.b = *reinterpret_cast<const float4*>(r0 + 16);
  return u.v;
}

// ---------------------------------------------------------------------------
// Kernel 1: scale + scaled series (past || future), one wave per batch row
// ---------------------------------------------------------------------------
__global__ void __launch_bounds__(256)
deepar_prep_kernel(const float* __restrict__ past_target,
                   const float* __restrict__ future_target,
                   float* __restrict__ series,   // [B][T_]
                   float* __restrict__ out) {    // writes scale section
  const int row  = blockIdx.x * 8 + (threadIdx.x >> 5);
  const int lane = threadIdx.x & 31;
  const float* pt = past_target + (size_t)row * C_;

  float s = 0.f;
  for (int k = lane; k < C_; k += 32) s += fabsf(pt[k]);
#pragma unroll
  for (int m = 16; m; m >>= 1) s += __shfl_xor(s, m, 32);
  const float scale = fmaxf(s * (1.f / C_), 1e-5f);

  if (lane == 0) out[1 + 2 * B_ * H_ + row] = scale;

  float* sr = series + (size_t)row * T_;
  for (int k = lane; k < C_; k += 32) sr[k] = pt[k] / scale;
  const float* ft = future_target + (size_t)row * H_;
  for (int k = lane; k < H_; k += 32) sr[C_ + k] = ft[k] / scale;
}

// ---------------------------------------------------------------------------
// Kernel 2: fused 2-layer LSTM recurrence (280 steps) + heads, WMMA f16->f32
// Block: 128 threads = 4 waves; 16 batch rows per block; wave t owns gate
// columns {g*64 + 16*t} for g=0..3 of both layers.
// ---------------------------------------------------------------------------
#define SMEM_F16 (G4_*32 + 3*G4_*64 + 16*32 + 2*16*64)         // weights + x + h bufs
#define SMEM_F32 (G4_ + G4_ + 64 + 64 + 16*E_ + 16)            // b0,b1,wmu,wsig,emb,red
#define SMEM_BYTES (SMEM_F16*2 + SMEM_F32*4)

__global__ void __launch_bounds__(128)
deepar_lstm_kernel(const float* __restrict__ ptf,  const float* __restrict__ ftf,
                   const int*   __restrict__ item, const float* __restrict__ past_age,
                   const float* __restrict__ future_age, const float* __restrict__ emb_table,
                   const float* __restrict__ Wih0, const float* __restrict__ Whh0,
                   const float* __restrict__ bih0, const float* __restrict__ bhh0,
                   const float* __restrict__ Wih1, const float* __restrict__ Whh1,
                   const float* __restrict__ bih1, const float* __restrict__ bhh1,
                   const float* __restrict__ Wmu,  const float* __restrict__ bmu,
                   const float* __restrict__ Wsig, const float* __restrict__ bsig,
                   const float* __restrict__ series,
                   float* __restrict__ out, float* __restrict__ partials) {
  extern __shared__ char smem[];
  _Float16* sWih0 = (_Float16*)smem;            // [256][32] (22 cols used, rest 0)
  _Float16* sWhh0 = sWih0 + G4_ * 32;           // [256][64]
  _Float16* sWih1 = sWhh0 + G4_ * 64;           // [256][64]
  _Float16* sWhh1 = sWih1 + G4_ * 64;           // [256][64]
  _Float16* sX    = sWhh1 + G4_ * 64;           // [16][32]
  _Float16* sH0   = sX    + 16 * 32;            // [16][64]
  _Float16* sH1   = sH0   + 16 * 64;            // [16][64]
  float* sB0   = (float*)(sH1 + 16 * 64);       // [256]
  float* sB1   = sB0  + G4_;                    // [256]
  float* sWmu  = sB1  + G4_;                    // [64]
  float* sWsig = sWmu + 64;                     // [64]
  float* sEmb  = sWsig + 64;                    // [16][E_]
  float* sRed  = sEmb + 16 * E_;                // [16]

  const int tid = threadIdx.x;
  const int wv  = tid >> 5;          // wave 0..3 -> gate-column quartet
  const int lane = tid & 31;
  const int l = lane & 15, hb = lane >> 4;
  const int rowBase = blockIdx.x * 16;

  // ---- stage weights to LDS (f16), biases/heads (f32) ----
  for (int i = tid; i < G4_ * 32; i += 128) sWih0[i] = (_Float16)0.f;
  __syncthreads();
  for (int i = tid; i < G4_ * 22; i += 128) {
    int r = i / 22, c = i % 22;
    sWih0[r * 32 + c] = (_Float16)Wih0[i];
  }
  for (int i = tid; i < G4_ * 64; i += 128) {
    sWhh0[i] = (_Float16)Whh0[i];
    sWih1[i] = (_Float16)Wih1[i];
    sWhh1[i] = (_Float16)Whh1[i];
  }
  for (int i = tid; i < G4_; i += 128) {
    sB0[i] = bih0[i] + bhh0[i];
    sB1[i] = bih1[i] + bhh1[i];
  }
  for (int i = tid; i < 64; i += 128) { sWmu[i] = Wmu[i]; sWsig[i] = Wsig[i]; }
  for (int i = tid; i < 16 * E_; i += 128) {
    int r = i / E_, e = i % E_;
    sEmb[i] = emb_table[(size_t)item[rowBase + r] * E_ + e];
  }
  for (int i = tid; i < 16 * 64; i += 128) { sH0[i] = (_Float16)0.f; sH1[i] = (_Float16)0.f; }
  __syncthreads();

  // ---- resident B-fragments for this wave's gate quartet ----
  v16h fWih0[4], fWhh0[4][2], fWih1[4][2], fWhh1[4][2];
#pragma unroll
  for (int g = 0; g < 4; ++g) {
    const int n0 = g * 64 + wv * 16;
    fWih0[g] = ld_frag16(sWih0 + n0 * 32, 32);
#pragma unroll
    for (int k = 0; k < 2; ++k) {
      fWhh0[g][k] = ld_frag16(sWhh0 + n0 * 64 + 32 * k, 64);
      fWih1[g][k] = ld_frag16(sWih1 + n0 * 64 + 32 * k, 64);
      fWhh1[g][k] = ld_frag16(sWhh1 + n0 * 64 + 32 * k, 64);
    }
  }
  float bb0[4], bb1[4];
#pragma unroll
  for (int g = 0; g < 4; ++g) {
    bb0[g] = sB0[g * 64 + wv * 16 + l];
    bb1[g] = sB1[g * 64 + wv * 16 + l];
  }

  v8f c0, c1;
#pragma unroll
  for (int r = 0; r < 8; ++r) { c0[r] = 0.f; c1[r] = 0.f; }

  float nllLocal = 0.f;

  for (int t = 0; t < T_; ++t) {
    // ---- build step-input feature tile (16 rows x 32 f16, 22 used) ----
    for (int idx = tid; idx < 16 * 32; idx += 128) {
      const int row = idx >> 5, col = idx & 31;
      const int gr = rowBase + row;
      const float* sr = series + (size_t)gr * T_;
      float v = 0.f;
      if (col == 0) {
        v = (t < C_) ? sr[t] : sr[t - 1];           // current (enc) / prev (dec)
      } else if (col <= NLAG) {
        const int tt = t - c_lags[col - 1];
        v = (tt >= 0) ? sr[tt] : 0.f;
      } else if (col <= NLAG + F_) {
        const int f = col - 1 - NLAG;
        v = (t < C_) ? ptf[((size_t)gr * C_ + t) * F_ + f]
                     : ftf[((size_t)gr * H_ + (t - C_)) * F_ + f];
      } else if (col <= NLAG + F_ + E_) {
        v = sEmb[row * E_ + (col - 1 - NLAG - F_)];
      } else if (col == NLAG + F_ + E_ + 1) {
        v = (t < C_) ? log1pf(past_age[(size_t)gr * C_ + t])
                     : log1pf(future_age[(size_t)gr * H_ + (t - C_)]);
      }
      sX[row * 32 + col] = (_Float16)v;
    }
    __syncthreads();                        // x ready; prev h visible

    const v16h aX   = ld_frag16(sX, 32);
    const v16h aH0a = ld_frag16(sH0, 64);
    const v16h aH0b = ld_frag16(sH0 + 32, 64);
    const v16h aH1a = ld_frag16(sH1, 64);
    const v16h aH1b = ld_frag16(sH1 + 32, 64);
    __syncthreads();                        // all prev-h reads done before overwrite

    // ---- layer 0: gates = x*Wih0^T + h0*Whh0^T + b0 ----
    v8f acc[4];
#pragma unroll
    for (int g = 0; g < 4; ++g) {
#pragma unroll
      for (int r = 0; r < 8; ++r) acc[g][r] = bb0[g];
      acc[g] = __builtin_amdgcn_wmma_f32_16x16x32_f16(false, aX,   false, fWih0[g],    (short)0, acc[g], false, false);
      acc[g] = __builtin_amdgcn_wmma_f32_16x16x32_f16(false, aH0a, false, fWhh0[g][0], (short)0, acc[g], false, false);
      acc[g] = __builtin_amdgcn_wmma_f32_16x16x32_f16(false, aH0b, false, fWhh0[g][1], (short)0, acc[g], false, false);
    }
#pragma unroll
    for (int r = 0; r < 8; ++r) {
      const float gi = fast_sigmoid(acc[0][r]);
      const float gf = fast_sigmoid(acc[1][r]);
      const float gg = fast_tanh(acc[2][r]);
      const float go = fast_sigmoid(acc[3][r]);
      const float cn = gf * c0[r] + gi * gg;
      c0[r] = cn;
      sH0[(r + 8 * hb) * 64 + wv * 16 + l] = (_Float16)(go * fast_tanh(cn));
    }
    __syncthreads();                        // h0(new) ready

    const v16h aH0na = ld_frag16(sH0, 64);
    const v16h aH0nb = ld_frag16(sH0 + 32, 64);

    // ---- layer 1: gates = h0new*Wih1^T + h1*Whh1^T + b1 ----
#pragma unroll
    for (int g = 0; g < 4; ++g) {
#pragma unroll
      for (int r = 0; r < 8; ++r) acc[g][r] = bb1[g];
      acc[g] = __builtin_amdgcn_wmma_f32_16x16x32_f16(false, aH0na, false, fWih1[g][0], (short)0, acc[g], false, false);
      acc[g] = __builtin_amdgcn_wmma_f32_16x16x32_f16(false, aH0nb, false, fWih1[g][1], (short)0, acc[g], false, false);
      acc[g] = __builtin_amdgcn_wmma_f32_16x16x32_f16(false, aH1a,  false, fWhh1[g][0], (short)0, acc[g], false, false);
      acc[g] = __builtin_amdgcn_wmma_f32_16x16x32_f16(false, aH1b,  false, fWhh1[g][1], (short)0, acc[g], false, false);
    }
#pragma unroll
    for (int r = 0; r < 8; ++r) {
      const float gi = fast_sigmoid(acc[0][r]);
      const float gf = fast_sigmoid(acc[1][r]);
      const float gg = fast_tanh(acc[2][r]);
      const float go = fast_sigmoid(acc[3][r]);
      const float cn = gf * c1[r] + gi * gg;
      c1[r] = cn;
      sH1[(r + 8 * hb) * 64 + wv * 16 + l] = (_Float16)(go * fast_tanh(cn));
    }

    // ---- decoder head ----
    if (t >= C_) {
      __syncthreads();                      // h1(new) complete
      if (tid < 16) {
        const int gr = rowBase + tid;
        const int jj = t - C_;
        float mu = bmu[0], sg = bsig[0];
        const _Float16* hr = sH1 + tid * 64;
#pragma unroll
        for (int j = 0; j < 64; ++j) {
          const float hv = (float)hr[j];
          mu = fmaf(hv, sWmu[j], mu);
          sg = fmaf(hv, sWsig[j], sg);
        }
        const float sigma = softplusf(sg);
        const float y = series[(size_t)gr * T_ + t];   // future_scaled[jj]
        const float z = (y - mu) / sigma;
        nllLocal += 0.91893853320467274f + logf(sigma) + 0.5f * z * z;
        out[1 + gr * H_ + jj] = mu;
        out[1 + B_ * H_ + gr * H_ + jj] = sigma;
      }
    }
  }

  // ---- deterministic per-block NLL partial ----
  if (tid < 16) sRed[tid] = nllLocal;
  __syncthreads();
  if (tid == 0) {
    float s = 0.f;
    for (int i = 0; i < 16; ++i) s += sRed[i];
    partials[blockIdx.x] = s;
  }
}

// ---------------------------------------------------------------------------
// Kernel 3: deterministic final loss reduction over 256 block partials
// ---------------------------------------------------------------------------
__global__ void __launch_bounds__(256)
deepar_finalize_kernel(const float* __restrict__ partials, float* __restrict__ out) {
  __shared__ float red[256];
  red[threadIdx.x] = partials[threadIdx.x];
  __syncthreads();
  for (int s = 128; s > 0; s >>= 1) {
    if (threadIdx.x < s) red[threadIdx.x] += red[threadIdx.x + s];
    __syncthreads();
  }
  if (threadIdx.x == 0) out[0] = red[0] / (float)(B_ * H_);
}

// ---------------------------------------------------------------------------
extern "C" void kernel_launch(void* const* d_in, const int* in_sizes, int n_in,
                              void* d_out, int out_size, void* d_ws, size_t ws_size,
                              hipStream_t stream) {
  const float* past_target   = (const float*)d_in[0];
  const float* ptf           = (const float*)d_in[1];
  const float* ftf           = (const float*)d_in[2];
  const float* future_target = (const float*)d_in[3];
  const int*   item          = (const int*)  d_in[4];
  const float* past_age      = (const float*)d_in[5];
  const float* future_age    = (const float*)d_in[6];
  const float* emb_table     = (const float*)d_in[7];
  const float* Wih0 = (const float*)d_in[8];
  const float* Whh0 = (const float*)d_in[9];
  const float* bih0 = (const float*)d_in[10];
  const float* bhh0 = (const float*)d_in[11];
  const float* Wih1 = (const float*)d_in[12];
  const float* Whh1 = (const float*)d_in[13];
  const float* bih1 = (const float*)d_in[14];
  const float* bhh1 = (const float*)d_in[15];
  const float* Wmu  = (const float*)d_in[16];
  const float* bmu  = (const float*)d_in[17];
  const float* Wsig = (const float*)d_in[18];
  const float* bsig = (const float*)d_in[19];
  float* out = (float*)d_out;

  // workspace carve: series[B][T] then block partials[256]
  float* wSeries = (float*)d_ws;
  float* wPart   = wSeries + (size_t)B_ * T_;

  (void)in_sizes; (void)n_in; (void)out_size; (void)ws_size;

  // raise dynamic LDS cap (123 KB; WGP has 320 KB) — idempotent, deterministic
  hipFuncSetAttribute(reinterpret_cast<const void*>(deepar_lstm_kernel),
                      hipFuncAttributeMaxDynamicSharedMemorySize, SMEM_BYTES);

  deepar_prep_kernel<<<B_ / 8, 256, 0, stream>>>(past_target, future_target, wSeries, out);

  deepar_lstm_kernel<<<B_ / 16, 128, SMEM_BYTES, stream>>>(
      ptf, ftf, item, past_age, future_age, emb_table,
      Wih0, Whh0, bih0, bhh0, Wih1, Whh1, bih1, bhh1,
      Wmu, bmu, Wsig, bsig, wSeries, out, wPart);

  deepar_finalize_kernel<<<1, 256, 0, stream>>>(wPart, out);
}
